// GCNGraphClassifier_64991445123484
// MI455X (gfx1250) — matrix operand order repaired and checked by reading, not compiled, not run on hardware
//
#include <hip/hip_runtime.h>

#define HIDF   128
#define NGRAPH 64
#define BN_EPS 1e-5f

typedef __attribute__((ext_vector_type(16))) __bf16         v16bf;
typedef __attribute__((ext_vector_type(8)))  float          v8f;
typedef __attribute__((ext_vector_type(8)))  unsigned short u16x8;

union BF16Frag { v16bf v; u16x8 h[2]; };

__device__ __forceinline__ unsigned short f2bf(float x) {
  unsigned int u = __float_as_uint(x);
  u += 0x7FFFu + ((u >> 16) & 1u);       // round-to-nearest-even
  return (unsigned short)(u >> 16);
}
__device__ __forceinline__ float bf2f(unsigned short h) {
  return __uint_as_float(((unsigned int)h) << 16);
}

// ---------------- small utility kernels ----------------

__global__ void k_fill0(float* __restrict__ p, int n) {
  int i = blockIdx.x * blockDim.x + threadIdx.x;
  if (i < n) p[i] = 0.0f;
}

__global__ void k_deg(const int* __restrict__ col, float* __restrict__ deg, int E) {
  int e = blockIdx.x * blockDim.x + threadIdx.x;
  if (e < E) atomicAdd(&deg[col[e]], 1.0f);
}

__global__ void k_dinv(const float* __restrict__ deg, float* __restrict__ dinv, int N) {
  int n = blockIdx.x * blockDim.x + threadIdx.x;
  if (n < N) dinv[n] = rsqrtf(deg[n] + 1.0f);   // +1 = self loop; deg >= 1 always
}

// Transpose W[K][C] -> WhT/WlT[C][K] split into bf16 hi/lo.
__global__ void k_prep_w(const float* __restrict__ W,
                         unsigned short* __restrict__ WhT,
                         unsigned short* __restrict__ WlT) {
  int gid = blockIdx.x * blockDim.x + threadIdx.x;
  if (gid >= HIDF * HIDF) return;
  int k = gid >> 7, c = gid & 127;
  float w = W[gid];
  unsigned short hi = f2bf(w);
  WhT[c * HIDF + k] = hi;
  WlT[c * HIDF + k] = f2bf(w - bf2f(hi));
}

// ---------------- WMMA GEMM: C[N,128] = act(A)[N,128] @ W[128,128] ----------------
// Optional fused BN+ReLU on A (act(x) = max(x*scale+shift, 0)).
// W supplied pre-transposed + bf16-split so B fragments are contiguous per lane.

__global__ __launch_bounds__(256) void k_gemm_wmma(
    const float* __restrict__ A,
    const unsigned short* __restrict__ WhT,
    const unsigned short* __restrict__ WlT,
    float* __restrict__ C,
    const float* __restrict__ scale,
    const float* __restrict__ shift,
    int applyBnRelu, int N)
{
  __shared__ __align__(16) unsigned short sAh[16 * HIDF];
  __shared__ __align__(16) unsigned short sAl[16 * HIDF];

  const int tid = threadIdx.x;
  const int rowBase = blockIdx.x * 16;

  // Stage 16x128 fp32 A tile -> split bf16 hi/lo in LDS (fused BN+ReLU here).
  for (int v = tid; v < 512; v += 256) {          // 512 float4 = 2048 elems
    int r = v >> 5;                               // row within tile
    int gr = rowBase + r; if (gr >= N) gr = N - 1;
    const float4* rowp = (const float4*)(A + (size_t)gr * HIDF);
    float4 x4 = rowp[v & 31];
    float xs[4] = {x4.x, x4.y, x4.z, x4.w};
    int base = v * 4;
#pragma unroll
    for (int j = 0; j < 4; ++j) {
      int idx = base + j;
      float x = xs[j];
      if (applyBnRelu) {
        int f = idx & 127;
        x = fmaxf(x * scale[f] + shift[f], 0.0f);
      }
      unsigned short hi = f2bf(x);
      sAh[idx] = hi;
      sAl[idx] = f2bf(x - bf2f(hi));
    }
  }
  __syncthreads();

  const int lane = tid & 31;
  const int wave = tid >> 5;            // n-tile 0..7
  const int m    = lane & 15;           // A row / output N within tile
  const int hs   = lane >> 4;           // half-wave select
  const int colN = wave * 16 + m;       // global output column

  v8f acc = {0.f, 0.f, 0.f, 0.f, 0.f, 0.f, 0.f, 0.f};

#pragma unroll
  for (int k0 = 0; k0 < HIDF; k0 += 32) {
    // A frag (16x32 bf16): lanes 0-15 K=[0..7]+[16..23], lanes 16-31 K=[8..15]+[24..31]
    int aoff = m * HIDF + k0 + hs * 8;
    BF16Frag ah, al, bh, bl;
    ah.h[0] = *(const u16x8*)(sAh + aoff);
    ah.h[1] = *(const u16x8*)(sAh + aoff + 16);
    al.h[0] = *(const u16x8*)(sAl + aoff);
    al.h[1] = *(const u16x8*)(sAl + aoff + 16);
    // B frag (32x16): lanes 0-15 K=[0..7]+[16..23], lanes 16-31 K=[8..15]+[24..31] (pre-transposed W)
    int boff = colN * HIDF + k0 + hs * 8;
    bh.h[0] = *(const u16x8*)(WhT + boff);
    bh.h[1] = *(const u16x8*)(WhT + boff + 16);
    bl.h[0] = *(const u16x8*)(WlT + boff);
    bl.h[1] = *(const u16x8*)(WlT + boff + 16);
    // bf16 hi/lo split product: AhBh + AhBl + AlBh
    acc = __builtin_amdgcn_wmma_f32_16x16x32_bf16(false, ah.v, false, bh.v,
                                                  (short)0, acc, false, false);
    acc = __builtin_amdgcn_wmma_f32_16x16x32_bf16(false, ah.v, false, bl.v,
                                                  (short)0, acc, false, false);
    acc = __builtin_amdgcn_wmma_f32_16x16x32_bf16(false, al.v, false, bh.v,
                                                  (short)0, acc, false, false);
  }

#pragma unroll
  for (int v = 0; v < 8; ++v) {
    int r = rowBase + v + hs * 8;       // C/D layout: M = v + 8*hs, N = lane%16
    if (r < N) C[(size_t)r * HIDF + colN] = acc[v];
  }
}

// ---------------- aggregation ----------------

// agg[n][f] = dinv[n]^2 * hW[n][f] + bias[f]   (self-loop + bias init)
__global__ void k_agg_init(const float* __restrict__ hW, const float* __restrict__ dinv,
                           const float* __restrict__ bias, float* __restrict__ agg, int N) {
  int gid = blockIdx.x * blockDim.x + threadIdx.x;
  if (gid >= N * HIDF) return;
  int n = gid >> 7, f = gid & 127;
  float d = dinv[n];
  agg[gid] = d * d * hW[gid] + bias[f];
}

// Edge scatter-add: agg[c] += dinv[r]*dinv[c]*hW[r] ; 4 threads per edge (32 feats each)
__global__ void k_agg_edges(const int* __restrict__ row, const int* __restrict__ col,
                            const float* __restrict__ dinv, const float* __restrict__ hW,
                            float* __restrict__ agg, int E) {
  int gid = blockIdx.x * blockDim.x + threadIdx.x;
  int e = gid >> 2, part = gid & 3;
  if (e >= E) return;
  int r = row[e], c = col[e];
  // hide gather latency: prefetch a lookahead edge's source row (global_prefetch_b8)
  if (part == 0) {
    int e2 = e + 4096;
    if (e2 < E) __builtin_prefetch(hW + (size_t)row[e2] * HIDF, 0, 3);
  }
  float norm = dinv[r] * dinv[c];
  const float4* src = (const float4*)(hW + (size_t)r * HIDF) + part * 8;
  float* dst = agg + (size_t)c * HIDF + part * 32;
#pragma unroll
  for (int i = 0; i < 8; ++i) {
    float4 v = src[i];
    atomicAdd(dst + i * 4 + 0, v.x * norm);
    atomicAdd(dst + i * 4 + 1, v.y * norm);
    atomicAdd(dst + i * 4 + 2, v.z * norm);
    atomicAdd(dst + i * 4 + 3, v.w * norm);
  }
}

// ---------------- batch norm ----------------

__global__ void k_bn_stats(const float* __restrict__ h, float* __restrict__ sums,
                           float* __restrict__ sumsq, int N) {
  __shared__ float s1[256], s2[256];
  int f = threadIdx.x & 127;
  int half = threadIdx.x >> 7;
  float a = 0.f, b = 0.f;
  for (int n = blockIdx.x * 2 + half; n < N; n += gridDim.x * 2) {
    float x = h[(size_t)n * HIDF + f];
    a += x; b += x * x;
  }
  s1[threadIdx.x] = a; s2[threadIdx.x] = b;
  __syncthreads();
  if (half == 0) {
    atomicAdd(&sums[f],  s1[f] + s1[f + 128]);
    atomicAdd(&sumsq[f], s2[f] + s2[f + 128]);
  }
}

__global__ void k_bn_final(const float* __restrict__ sums, const float* __restrict__ sumsq,
                           const float* __restrict__ gamma, const float* __restrict__ beta,
                           float* __restrict__ scale, float* __restrict__ shift, int N) {
  int f = threadIdx.x;
  if (f >= HIDF) return;
  float inv_n = 1.0f / (float)N;
  float mu  = sums[f] * inv_n;
  float var = sumsq[f] * inv_n - mu * mu;
  float sc  = gamma[f] * rsqrtf(var + BN_EPS);
  scale[f] = sc;
  shift[f] = beta[f] - mu * sc;
}

// ---------------- pooling + head ----------------

__global__ void k_pool(const float* __restrict__ h, const float* __restrict__ scale,
                       const float* __restrict__ shift, const int* __restrict__ batch,
                       float* __restrict__ psum, float* __restrict__ pcnt, int N) {
  int gid = blockIdx.x * blockDim.x + threadIdx.x;
  if (gid >= N * HIDF) return;
  int n = gid >> 7, f = gid & 127;
  float y = fmaxf(h[gid] * scale[f] + shift[f], 0.0f);   // BN2 + ReLU fused
  int g = batch[n];
  atomicAdd(&psum[g * HIDF + f], y);
  if (f == 0) atomicAdd(&pcnt[g], 1.0f);
}

__global__ void k_head(const float* __restrict__ psum, const float* __restrict__ pcnt,
                       const float* __restrict__ W1, const float* __restrict__ b1,
                       const float* __restrict__ W2, const float* __restrict__ b2,
                       float* __restrict__ out) {
  __shared__ float pooled[NGRAPH * HIDF];   // 32 KB
  __shared__ float z[NGRAPH * 64];          // 16 KB
  for (int i = threadIdx.x; i < NGRAPH * HIDF; i += blockDim.x) {
    int g = i >> 7;
    pooled[i] = psum[i] / fmaxf(pcnt[g], 1.0f);
  }
  __syncthreads();
  for (int i = threadIdx.x; i < NGRAPH * 64; i += blockDim.x) {
    int g = i >> 6, j = i & 63;
    float acc = b1[j];
#pragma unroll 4
    for (int k = 0; k < HIDF; ++k) acc += pooled[g * HIDF + k] * W1[k * 64 + j];
    z[i] = fmaxf(acc, 0.0f);
  }
  __syncthreads();
  for (int i = threadIdx.x; i < NGRAPH * 2; i += blockDim.x) {
    int g = i >> 1, c = i & 1;
    float acc = b2[c];
#pragma unroll 8
    for (int j = 0; j < 64; ++j) acc += z[g * 64 + j] * W2[j * 2 + c];
    out[i] = acc;
  }
}

// ---------------- host launcher ----------------

extern "C" void kernel_launch(void* const* d_in, const int* in_sizes, int n_in,
                              void* d_out, int out_size, void* d_ws, size_t ws_size,
                              hipStream_t stream) {
  const float* x      = (const float*)d_in[0];
  const int*   edge   = (const int*)d_in[1];    // [2, E]
  const int*   batch  = (const int*)d_in[2];
  const float* W1     = (const float*)d_in[3];
  const float* b1     = (const float*)d_in[4];
  const float* W2     = (const float*)d_in[5];
  const float* b2     = (const float*)d_in[6];
  const float* gamma1 = (const float*)d_in[7];
  const float* beta1  = (const float*)d_in[8];
  const float* gamma2 = (const float*)d_in[9];
  const float* beta2  = (const float*)d_in[10];
  const float* lin1W  = (const float*)d_in[11];
  const float* lin1b  = (const float*)d_in[12];
  const float* lin2W  = (const float*)d_in[13];
  const float* lin2b  = (const float*)d_in[14];
  float* out = (float*)d_out;

  const int N = in_sizes[0] / HIDF;
  const int E = in_sizes[1] / 2;

  char* wsb = (char*)d_ws;
  size_t off = 0;
  auto carve = [&](size_t bytes) -> void* {
    off = (off + 255) & ~(size_t)255;
    void* p = wsb + off;
    off += bytes;
    return p;
  };
  float* deg   = (float*)carve((size_t)N * 4);
  float* dinv  = (float*)carve((size_t)N * 4);
  float* hW    = (float*)carve((size_t)N * HIDF * 4);
  float* agg   = (float*)carve((size_t)N * HIDF * 4);
  unsigned short* Wh1 = (unsigned short*)carve(HIDF * HIDF * 2);
  unsigned short* Wl1 = (unsigned short*)carve(HIDF * HIDF * 2);
  unsigned short* Wh2 = (unsigned short*)carve(HIDF * HIDF * 2);
  unsigned short* Wl2 = (unsigned short*)carve(HIDF * HIDF * 2);
  float* bnstat = (float*)carve(256 * 4);      // sums[128] | sumsq[128]
  float* bnsum  = bnstat;
  float* bnsq   = bnstat + 128;
  float* scale  = (float*)carve(HIDF * 4);
  float* shift  = (float*)carve(HIDF * 4);
  float* pool   = (float*)carve((NGRAPH * HIDF + NGRAPH) * 4);  // psum | pcnt
  float* psum   = pool;
  float* pcnt   = pool + NGRAPH * HIDF;

  const int NF = N * HIDF;
  const int gN   = (N + 255) / 256;
  const int gE   = (E + 255) / 256;
  const int gE4  = (E * 4 + 255) / 256;
  const int gNF  = (NF + 255) / 256;
  const int gT   = (N + 15) / 16;

  // degrees / normalization
  k_fill0<<<gN, 256, 0, stream>>>(deg, N);
  k_deg<<<gE, 256, 0, stream>>>(edge + E, deg, E);
  k_dinv<<<gN, 256, 0, stream>>>(deg, dinv, N);

  // weight prep (transpose + bf16 hi/lo split)
  k_prep_w<<<(HIDF * HIDF + 255) / 256, 256, 0, stream>>>(W1, Wh1, Wl1);
  k_prep_w<<<(HIDF * HIDF + 255) / 256, 256, 0, stream>>>(W2, Wh2, Wl2);

  // ---- layer 1 ----
  k_gemm_wmma<<<gT, 256, 0, stream>>>(x, Wh1, Wl1, hW, nullptr, nullptr, 0, N);
  k_agg_init<<<gNF, 256, 0, stream>>>(hW, dinv, b1, agg, N);
  k_agg_edges<<<gE4, 256, 0, stream>>>(edge, edge + E, dinv, hW, agg, E);
  k_fill0<<<1, 256, 0, stream>>>(bnstat, 256);
  k_bn_stats<<<256, 256, 0, stream>>>(agg, bnsum, bnsq, N);
  k_bn_final<<<1, 128, 0, stream>>>(bnsum, bnsq, gamma1, beta1, scale, shift, N);

  // ---- layer 2 (BN1+ReLU fused into GEMM A staging) ----
  k_gemm_wmma<<<gT, 256, 0, stream>>>(agg, Wh2, Wl2, hW, scale, shift, 1, N);
  k_agg_init<<<gNF, 256, 0, stream>>>(hW, dinv, b2, agg, N);
  k_agg_edges<<<gE4, 256, 0, stream>>>(edge, edge + E, dinv, hW, agg, E);
  k_fill0<<<1, 256, 0, stream>>>(bnstat, 256);
  k_bn_stats<<<256, 256, 0, stream>>>(agg, bnsum, bnsq, N);
  k_bn_final<<<1, 128, 0, stream>>>(bnsum, bnsq, gamma2, beta2, scale, shift, N);

  // ---- pool (BN2+ReLU fused) + head ----
  k_fill0<<<(NGRAPH * HIDF + NGRAPH + 255) / 256, 256, 0, stream>>>(pool, NGRAPH * HIDF + NGRAPH);
  k_pool<<<gNF, 256, 0, stream>>>(agg, scale, shift, batch, psum, pcnt, N);
  k_head<<<1, 256, 0, stream>>>(psum, pcnt, lin1W, lin1b, lin2W, lin2b, out);
}
